// EnhancedGAT_50929722196348
// MI455X (gfx1250) — matrix operand order repaired and checked by reading, not compiled, not run on hardware
//
#include <hip/hip_runtime.h>

typedef __attribute__((ext_vector_type(16))) _Float16 v16h;
typedef __attribute__((ext_vector_type(8)))  _Float16 v8h;
typedef __attribute__((ext_vector_type(8)))  float    v8f;

#define HIDC 32

__device__ __forceinline__ unsigned f2key(float f) {
  unsigned b = __float_as_uint(f);
  return (b & 0x80000000u) ? ~b : (b | 0x80000000u);   // monotone float->uint map
}
__device__ __forceinline__ float key2f(unsigned k) {
  return __uint_as_float((k & 0x80000000u) ? (k & 0x7FFFFFFFu) : ~k);
}
__device__ __forceinline__ float lrelu(float x) { return x >= 0.f ? x : 0.2f * x; }

// ---------------- edge encoder: edge_emb = edge_attr @ We_enc + be_enc ----------------
__global__ void k_edge_enc(const float* __restrict__ ea, const float* __restrict__ W,
                           const float* __restrict__ b, float* __restrict__ out, int E) {
  int t = blockIdx.x * blockDim.x + threadIdx.x;
  if (t >= E * HIDC) return;
  int e = t / HIDC, c = t % HIDC;
  float s = b[c];
#pragma unroll
  for (int j = 0; j < 4; ++j) s += ea[e * 4 + j] * W[j * HIDC + c];
  out[t] = s;
}

// ---------------- incoming degree ----------------
__global__ void k_cnt(const int* __restrict__ dst, float* __restrict__ cnt, int E) {
  int t = blockIdx.x * blockDim.x + threadIdx.x;
  if (t < E) atomicAdd(&cnt[dst[t]], 1.f);
}

// ---------------- weight convert + transpose (zero-padded K) -> f16 [O, Kpad] ----------------
__global__ void k_wt(const float* __restrict__ W, _Float16* __restrict__ Wt,
                     int K, int O, int Kpad) {
  int t = blockIdx.x * blockDim.x + threadIdx.x;
  if (t >= O * Kpad) return;
  int o = t / Kpad, k = t % Kpad;
  Wt[t] = (k < K) ? (_Float16)W[k * O + o] : (_Float16)0.f;
}

// ---------------- activation convert (zero-padded K) -> f16 [M, Kpad] ----------------
__global__ void k_cvtA(const float* __restrict__ A, _Float16* __restrict__ A16,
                       int M, int K, int Kpad) {
  int t = blockIdx.x * blockDim.x + threadIdx.x;
  if (t >= M * Kpad) return;
  int m = t / Kpad, k = t % Kpad;
  A16[t] = (k < K) ? (_Float16)A[(size_t)m * K + k] : (_Float16)0.f;
}

// ---------------- WMMA GEMM: C[M,O] = A16[M,Kpad] x Bt[O,Kpad]^T (both f16, padded)
// one wave per (16 x CT*16) C strip: A fragment loaded once, CT WMMAs per k-step.
// Fragment k order: contiguous, k = 16*(lane/16) + s in both A and B (wave32). No divergence in body.
template <int CT>
__global__ void k_gemm(const _Float16* __restrict__ A16, const _Float16* __restrict__ Bt,
                       float* __restrict__ C, int M, int Kpad, int O) {
  int wave = blockIdx.x * (blockDim.x >> 5) + (threadIdx.x >> 5);
  int lane = threadIdx.x & 31;
  int cgroups = (O >> 4) / CT;
  int rtiles = (M + 15) >> 4;
  if (wave >= rtiles * cgroups) return;            // wave-uniform exit
  int rt = wave / cgroups, cg = wave % cgroups;
  int lrow = rt * 16 + (lane & 15);
  int row = lrow < M ? lrow : M - 1;               // clamp loads, guard stores (no divergence)
  int colb = (lane & 15);
  bool hi = lane >= 16;
  v8f acc[CT];
#pragma unroll
  for (int j = 0; j < CT; ++j) acc[j] = {};

  for (int k0 = 0; k0 < Kpad; k0 += 32) {
    // A 16x32 f16: lanes 0-15 hold K k0..k0+15; lanes 16-31 hold K k0+16..k0+31
    int kA = k0 + (hi ? 16 : 0);
    const _Float16* ap = A16 + (size_t)row * Kpad + kA;
    v8h alo = *(const v8h*)ap;          // K kA..kA+7   -> global_load_b128
    v8h ahi = *(const v8h*)(ap + 8);    // K kA+8..+15  -> global_load_b128
    v16h a = __builtin_shufflevector(alo, ahi, 0, 1, 2, 3, 4, 5, 6, 7,
                                     8, 9, 10, 11, 12, 13, 14, 15);
    // B 32x16 f16: lanes 0-15 = K k0..k0+15, lanes 16-31 = K k0+16..k0+31 (col = lane&15)
    int kB = k0 + (hi ? 16 : 0);
#pragma unroll
    for (int j = 0; j < CT; ++j) {
      int col = (cg * CT + j) * 16 + colb;
      v16h b = *(const v16h*)(Bt + (size_t)col * Kpad + kB);  // 2x global_load_b128
      acc[j] = __builtin_amdgcn_wmma_f32_16x16x32_f16(false, a, false, b, (short)0,
                                                      acc[j], false, false);
    }
  }
  // C/D f32 16x16: VGPR r -> M = rt*16 + r (+8 for hi half), N = lane&15
  int mbase = rt * 16 + (hi ? 8 : 0);
#pragma unroll
  for (int j = 0; j < CT; ++j) {
    int col = (cg * CT + j) * 16 + colb;
#pragma unroll
    for (int r = 0; r < 8; ++r) {
      int m = mbase + r;
      if (m < M) C[(size_t)m * O + col] = acc[j][r];
    }
  }
}

// ---------------- per-node attention scores a = sum_c h[n,h,c]*att[h,c] ----------------
__global__ void k_att(const float* __restrict__ h, const float* __restrict__ att,
                      float* __restrict__ out, int N, int H) {
  int t = blockIdx.x * blockDim.x + threadIdx.x;
  if (t >= N * H) return;
  int n = t / H, hh = t % H;
  const float* hp = h + (size_t)n * H * HIDC + hh * HIDC;
  const float* ap = att + hh * HIDC;
  float s = 0.f;
#pragma unroll
  for (int c = 0; c < HIDC; ++c) s += hp[c] * ap[c];
  out[t] = s;
}

// ---------------- v_e[h] = We[:, h*32:(h+1)*32] @ ae[h]  (linearity trick) ----------------
__global__ void k_ve(const float* __restrict__ We, const float* __restrict__ ae,
                     float* __restrict__ ve, int H) {
  int t = threadIdx.x;
  if (t >= H * HIDC) return;
  int hh = t / HIDC, k = t % HIDC;
  int O = H * HIDC;
  float s = 0.f;
#pragma unroll
  for (int c = 0; c < HIDC; ++c) s += We[k * O + hh * HIDC + c] * ae[hh * HIDC + c];
  ve[t] = s;
}

// ---------------- per-edge edge-feature alpha term + segment-sum for loop mean ----------------
__global__ void k_aee(const float* __restrict__ ee, const float* __restrict__ ve,
                      const int* __restrict__ dst, float* __restrict__ aee,
                      float* __restrict__ esum, int E, int H) {
  int t = blockIdx.x * blockDim.x + threadIdx.x;
  if (t >= E * H) return;
  int e = t / H, hh = t % H;
  const float* ep = ee + (size_t)e * HIDC;
  const float* vp = ve + hh * HIDC;
  float s = 0.f;
#pragma unroll
  for (int c = 0; c < HIDC; ++c) s += ep[c] * vp[c];
  aee[t] = s;
  atomicAdd(&esum[dst[e] * H + hh], s);
}

// ---------------- raw edge alpha + segment max ----------------
__global__ void k_eam(const int* __restrict__ src, const int* __restrict__ dst,
                      const float* __restrict__ as_, const float* __restrict__ ad_,
                      const float* __restrict__ aee, float* __restrict__ al,
                      unsigned* __restrict__ amax, int E, int H) {
  int t = blockIdx.x * blockDim.x + threadIdx.x;
  if (t >= E * H) return;
  int e = t / H, hh = t % H;
  float raw = as_[src[e] * H + hh] + ad_[dst[e] * H + hh];
  if (aee) raw += aee[t];
  raw = lrelu(raw);
  al[t] = raw;
  atomicMax(&amax[dst[e] * H + hh], f2key(raw));
}

// ---------------- self-loop alpha (mean edge feature via esum/deg) + segment max ----------------
__global__ void k_nlam(const float* __restrict__ as_, const float* __restrict__ ad_,
                       const float* __restrict__ esum, const float* __restrict__ cnt,
                       float* __restrict__ aloop, unsigned* __restrict__ amax, int N, int H) {
  int t = blockIdx.x * blockDim.x + threadIdx.x;
  if (t >= N * H) return;
  int n = t / H;
  float raw = as_[t] + ad_[t];
  if (esum) raw += esum[t] / fmaxf(cnt[n], 1.f);
  raw = lrelu(raw);
  aloop[t] = raw;
  atomicMax(&amax[t], f2key(raw));
}

// ---------------- exp(alpha - max) + denominator ----------------
__global__ void k_eex(const int* __restrict__ dst, float* __restrict__ al,
                      const unsigned* __restrict__ amax, float* __restrict__ den,
                      int E, int H) {
  int t = blockIdx.x * blockDim.x + threadIdx.x;
  if (t >= E * H) return;
  int e = t / H, hh = t % H;
  float ex = __expf(al[t] - key2f(amax[dst[e] * H + hh]));
  al[t] = ex;
  atomicAdd(&den[dst[e] * H + hh], ex);
}

__global__ void k_nex(float* __restrict__ aloop, const unsigned* __restrict__ amax,
                      float* __restrict__ den, int N, int H) {
  int t = blockIdx.x * blockDim.x + threadIdx.x;
  if (t >= N * H) return;
  float ex = __expf(aloop[t] - key2f(amax[t]));
  aloop[t] = ex;
  atomicAdd(&den[t], ex);
}

// ---------------- scatter aggregation acc[dst] += ex * h_lin[src] ----------------
__global__ void k_agg(const int* __restrict__ src, const int* __restrict__ dst,
                      const float* __restrict__ ex, const float* __restrict__ hlin,
                      float* __restrict__ acc, int E, int H) {
  int t = blockIdx.x * blockDim.x + threadIdx.x;
  int O = H * HIDC;
  if (t >= E * O) return;
  int e = t / O, c = t % O;
  int hh = c / HIDC;
  atomicAdd(&acc[(size_t)dst[e] * O + c], ex[e * H + hh] * hlin[(size_t)src[e] * O + c]);
}

// ---------------- finalize: add self-loop term, divide by den, bias, relu ----------------
__global__ void k_fin(const float* __restrict__ acc, const float* __restrict__ hlin,
                      const float* __restrict__ exloop, const float* __restrict__ den,
                      const float* __restrict__ bias, float* __restrict__ out,
                      int N, int H, int do_relu) {
  int t = blockIdx.x * blockDim.x + threadIdx.x;
  int O = H * HIDC;
  if (t >= N * O) return;
  int n = t / O, c = t % O;
  int hh = c / HIDC;
  float v = (acc[t] + exloop[n * H + hh] * hlin[t]) / (den[n * H + hh] + 1e-16f) + bias[c];
  if (do_relu) v = fmaxf(v, 0.f);
  out[t] = v;
}

// ---------------- global mean pool ----------------
__global__ void k_pool(const float* __restrict__ h, const int* __restrict__ batch,
                       float* __restrict__ gsum, float* __restrict__ gcnt, int N) {
  int t = blockIdx.x * blockDim.x + threadIdx.x;
  if (t >= N * HIDC) return;
  int n = t / HIDC, c = t % HIDC;
  int g = batch[n];
  atomicAdd(&gsum[g * HIDC + c], h[t]);
  if (c == 0) atomicAdd(&gcnt[g], 1.f);
}

// ---------------- descriptor MLP ----------------
__global__ void k_desc(const float* __restrict__ d, const float* __restrict__ Wd,
                       const float* __restrict__ bd, float* __restrict__ out, int G) {
  int t = blockIdx.x * blockDim.x + threadIdx.x;
  if (t >= G * HIDC) return;
  int g = t / HIDC, c = t % HIDC;
  float s = bd[c];
  for (int j = 0; j < 48; ++j) s += d[g * 48 + j] * Wd[j * HIDC + c];
  out[t] = fmaxf(s, 0.f);
}

// ---------------- head: sigmoid([graph_emb, desc_emb] @ Wl + bl) ----------------
__global__ void k_head(const float* __restrict__ gsum, const float* __restrict__ gcnt,
                       const float* __restrict__ desc, const float* __restrict__ Wl,
                       const float* __restrict__ bl, float* __restrict__ out, int G) {
  int g = blockIdx.x * blockDim.x + threadIdx.x;
  if (g >= G) return;
  float inv = 1.f / fmaxf(gcnt[g], 1.f);
  float s = bl[0];
#pragma unroll
  for (int c = 0; c < HIDC; ++c) s += (gsum[g * HIDC + c] * inv) * Wl[c];
#pragma unroll
  for (int c = 0; c < HIDC; ++c) s += desc[g * HIDC + c] * Wl[HIDC + c];
  out[g] = 1.f / (1.f + __expf(-s));
}

static inline int nb(long long n, int b) { return (int)((n + b - 1) / b); }

extern "C" void kernel_launch(void* const* d_in, const int* in_sizes, int n_in,
                              void* d_out, int out_size, void* d_ws, size_t ws_size,
                              hipStream_t stream) {
  const float* x      = (const float*)d_in[0];
  const int*   ei     = (const int*)d_in[1];
  const float* eattr  = (const float*)d_in[2];
  const int*   batch  = (const int*)d_in[3];
  const float* descs  = (const float*)d_in[4];
  const float* We_enc = (const float*)d_in[5];
  const float* be_enc = (const float*)d_in[6];
  const float* W1  = (const float*)d_in[7];
  const float* as1 = (const float*)d_in[8];
  const float* ad1 = (const float*)d_in[9];
  const float* b1  = (const float*)d_in[10];
  const float* W2  = (const float*)d_in[11];
  const float* We2 = (const float*)d_in[12];
  const float* as2 = (const float*)d_in[13];
  const float* ad2 = (const float*)d_in[14];
  const float* ae2 = (const float*)d_in[15];
  const float* b2  = (const float*)d_in[16];
  const float* W3  = (const float*)d_in[17];
  const float* We3 = (const float*)d_in[18];
  const float* as3 = (const float*)d_in[19];
  const float* ad3 = (const float*)d_in[20];
  const float* ae3 = (const float*)d_in[21];
  const float* b3  = (const float*)d_in[22];
  const float* W4  = (const float*)d_in[23];
  const float* We4 = (const float*)d_in[24];
  const float* as4 = (const float*)d_in[25];
  const float* ad4 = (const float*)d_in[26];
  const float* ae4 = (const float*)d_in[27];
  const float* b4  = (const float*)d_in[28];
  const float* Wd  = (const float*)d_in[29];
  const float* bd  = (const float*)d_in[30];
  const float* Wl  = (const float*)d_in[31];
  const float* bl  = (const float*)d_in[32];
  float* out = (float*)d_out;

  const int Nn = in_sizes[0] / 8;
  const int Ee = in_sizes[1] / 2;
  const int Gg = in_sizes[4] / 48;
  const int* srcp = ei;
  const int* dstp = ei + Ee;

  // ---- workspace layout ----
  char* ws = (char*)d_ws;
  size_t off = 0;
  auto alloc = [&](size_t bytes) -> char* {
    char* p = ws + off;
    off += (bytes + 255) & ~(size_t)255;
    return p;
  };
  float*    f_ee    = (float*)alloc((size_t)Ee * HIDC * 4);   // edge embeddings
  float*    f_hA    = (float*)alloc((size_t)Nn * 128 * 4);    // layer activations (ping)
  float*    f_hB    = (float*)alloc((size_t)Nn * 128 * 4);    // h_lin (GEMM output)
  float*    f_acc   = (float*)alloc((size_t)Nn * 128 * 4);    // aggregation accumulator
  float*    f_h4    = (float*)alloc((size_t)Nn * HIDC * 4);   // layer-4 output
  _Float16* f_A16   = (_Float16*)alloc((size_t)Nn * 128 * 2); // f16 GEMM activations
  _Float16* f_Wt    = (_Float16*)alloc(128 * 128 * 2);        // transposed f16 weights
  float*    f_ve    = (float*)alloc(4 * HIDC * 4);
  float*    f_as    = (float*)alloc((size_t)Nn * 4 * 4);
  float*    f_ad    = (float*)alloc((size_t)Nn * 4 * 4);
  float*    f_aee   = (float*)alloc((size_t)Ee * 4 * 4);
  float*    f_al    = (float*)alloc((size_t)Ee * 4 * 4);
  float*    f_esum  = (float*)alloc((size_t)Nn * 4 * 4);
  float*    f_cnt   = (float*)alloc((size_t)Nn * 4);
  unsigned* f_amax  = (unsigned*)alloc((size_t)Nn * 4 * 4);
  float*    f_den   = (float*)alloc((size_t)Nn * 4 * 4);
  float*    f_aloop = (float*)alloc((size_t)Nn * 4 * 4);
  float*    f_gsum  = (float*)alloc((size_t)Gg * HIDC * 4);
  float*    f_gcnt  = (float*)alloc((size_t)Gg * 4);
  float*    f_desc  = (float*)alloc((size_t)Gg * HIDC * 4);
  if (off > ws_size) return;  // insufficient scratch; deterministic no-op

  // ---- once: edge embeddings + incoming degree ----
  k_edge_enc<<<nb((long long)Ee * HIDC, 256), 256, 0, stream>>>(eattr, We_enc, be_enc, f_ee, Ee);
  hipMemsetAsync(f_cnt, 0, (size_t)Nn * 4, stream);
  k_cnt<<<nb(Ee, 256), 256, 0, stream>>>(dstp, f_cnt, Ee);

  // ---- one GAT layer ----
  auto gat = [&](const float* xin, int K, const float* W, const float* as_p,
                 const float* ad_p, const float* bias, int H, int do_relu,
                 const float* We, const float* ae_p, float* outp) {
    int O = H * HIDC;
    int Kpad = (K + 31) & ~31;
    k_wt<<<nb((long long)O * Kpad, 256), 256, 0, stream>>>(W, f_Wt, K, O, Kpad);
    k_cvtA<<<nb((long long)Nn * Kpad, 256), 256, 0, stream>>>(xin, f_A16, Nn, K, Kpad);
    int rtiles = (Nn + 15) / 16;
    if (((O >> 4) & 3) == 0) {
      int waves = rtiles * ((O >> 4) / 4);
      k_gemm<4><<<nb(waves, 8), 256, 0, stream>>>(f_A16, f_Wt, f_hB, Nn, Kpad, O);
    } else {
      int waves = rtiles * ((O >> 4) / 2);
      k_gemm<2><<<nb(waves, 8), 256, 0, stream>>>(f_A16, f_Wt, f_hB, Nn, Kpad, O);
    }
    k_att<<<nb((long long)Nn * H, 256), 256, 0, stream>>>(f_hB, as_p, f_as, Nn, H);
    k_att<<<nb((long long)Nn * H, 256), 256, 0, stream>>>(f_hB, ad_p, f_ad, Nn, H);
    if (We) {
      k_ve<<<1, H * HIDC, 0, stream>>>(We, ae_p, f_ve, H);
      hipMemsetAsync(f_esum, 0, (size_t)Nn * H * 4, stream);
      k_aee<<<nb((long long)Ee * H, 256), 256, 0, stream>>>(f_ee, f_ve, dstp, f_aee, f_esum, Ee, H);
    }
    hipMemsetAsync(f_amax, 0, (size_t)Nn * H * 4, stream);
    hipMemsetAsync(f_den, 0, (size_t)Nn * H * 4, stream);
    hipMemsetAsync(f_acc, 0, (size_t)Nn * O * 4, stream);
    k_eam<<<nb((long long)Ee * H, 256), 256, 0, stream>>>(srcp, dstp, f_as, f_ad,
                                                          We ? f_aee : (const float*)nullptr,
                                                          f_al, f_amax, Ee, H);
    k_nlam<<<nb((long long)Nn * H, 256), 256, 0, stream>>>(f_as, f_ad,
                                                           We ? f_esum : (const float*)nullptr,
                                                           f_cnt, f_aloop, f_amax, Nn, H);
    k_eex<<<nb((long long)Ee * H, 256), 256, 0, stream>>>(dstp, f_al, f_amax, f_den, Ee, H);
    k_nex<<<nb((long long)Nn * H, 256), 256, 0, stream>>>(f_aloop, f_amax, f_den, Nn, H);
    k_agg<<<nb((long long)Ee * O, 256), 256, 0, stream>>>(srcp, dstp, f_al, f_hB, f_acc, Ee, H);
    k_fin<<<nb((long long)Nn * O, 256), 256, 0, stream>>>(f_acc, f_hB, f_aloop, f_den, bias,
                                                          outp, Nn, H, do_relu);
  };

  gat(x,    8,   W1, as1, ad1, b1, 4, 1, nullptr, nullptr, f_hA);
  gat(f_hA, 128, W2, as2, ad2, b2, 4, 1, We2, ae2, f_hA);
  gat(f_hA, 128, W3, as3, ad3, b3, 4, 1, We3, ae3, f_hA);
  gat(f_hA, 128, W4, as4, ad4, b4, 1, 0, We4, ae4, f_h4);

  // ---- pooling + descriptor MLP + head ----
  hipMemsetAsync(f_gsum, 0, (size_t)Gg * HIDC * 4, stream);
  hipMemsetAsync(f_gcnt, 0, (size_t)Gg * 4, stream);
  k_pool<<<nb((long long)Nn * HIDC, 256), 256, 0, stream>>>(f_h4, batch, f_gsum, f_gcnt, Nn);
  k_desc<<<nb((long long)Gg * HIDC, 256), 256, 0, stream>>>(descs, Wd, bd, f_desc, Gg);
  k_head<<<nb(Gg, 256), 256, 0, stream>>>(f_gsum, f_gcnt, f_desc, Wl, bl, out, Gg);
}
